// ThinkerDecoderLayer_60610578481162
// MI455X (gfx1250) — matrix operand order repaired and checked by reading, not compiled, not run on hardware
//
#include <hip/hip_runtime.h>

// ---------------------------------------------------------------------------
// Types / WMMA helpers (CDNA5 gfx1250, wave32)
// ---------------------------------------------------------------------------
typedef __bf16 bf16;
typedef bf16  bf16x16 __attribute__((ext_vector_type(16)));
typedef float f32x8   __attribute__((ext_vector_type(8)));

#define WMMA_BF16(a, b, c) \
  __builtin_amdgcn_wmma_f32_16x16x32_bf16(false, (a), false, (b), (short)0, (c), false, false)

union FragU { bf16x16 v; uint4 u[2]; };

// A-fragment (16x32 bf16, row-major source with element stride `ld`):
// lane L: m = L&15 ; K-halves {0..7,16..23} (lanes 0-15) / {8..15,24..31} (lanes 16-31)
__device__ __forceinline__ bf16x16 load_a_frag(const bf16* base, int ld, int lane) {
  const bf16* p = base + (lane & 15) * ld + ((lane >> 4) << 3);
  FragU f;
  f.u[0] = *(const uint4*)(p);
  f.u[1] = *(const uint4*)(p + 16);
  return f.v;
}

// B-fragment (32x16 bf16) read from a K-contiguous (i.e. B^T [n][k]) tile:
// lane L: n = L&15 ; K-halves {0..7,16..23} (lanes 0-15) / {8..15,24..31} (lanes 16-31)
__device__ __forceinline__ bf16x16 load_b_frag(const bf16* base, int ld, int lane) {
  const bf16* p = base + (lane & 15) * ld + ((lane >> 4) << 3);
  FragU f;
  f.u[0] = *(const uint4*)(p);
  f.u[1] = *(const uint4*)(p + 16);
  return f.v;
}

// ---------------------------------------------------------------------------
// Problem constants
// ---------------------------------------------------------------------------
#define BSZ   2
#define TSEQ  2048
#define DMOD  2048
#define NH    16
#define NKV   4
#define HD    128
#define INTER 8192
#define MTOK  (BSZ * TSEQ)   // 4096 token rows

// ---------------------------------------------------------------------------
// Weight transpose + fp32 -> bf16 convert:  W[K][N] f32  ->  Wt[N][K] bf16
// ---------------------------------------------------------------------------
__global__ __launch_bounds__(256) void transpose_cvt_kernel(
    const float* __restrict__ W, bf16* __restrict__ Wt, int K, int N) {
  __shared__ float tile[32][33];
  int nb = blockIdx.x * 32, kb = blockIdx.y * 32;
  int tx = threadIdx.x & 31, ty = threadIdx.x >> 5;   // 32 x 8
#pragma unroll
  for (int i = 0; i < 32; i += 8)
    tile[ty + i][tx] = W[(size_t)(kb + ty + i) * N + nb + tx];
  __syncthreads();
#pragma unroll
  for (int i = 0; i < 32; i += 8)
    Wt[(size_t)(nb + ty + i) * K + kb + tx] = (bf16)tile[tx][ty + i];
}

// ---------------------------------------------------------------------------
// RMSNorm (f32 in) -> bf16 out, one block per row, D = 2048
// ---------------------------------------------------------------------------
__global__ __launch_bounds__(256) void rmsnorm_kernel(
    const float* __restrict__ x, const float* __restrict__ w,
    bf16* __restrict__ y) {
  int row = blockIdx.x, tid = threadIdx.x;
  const float* xr = x + (size_t)row * DMOD;
  float s = 0.f;
#pragma unroll
  for (int i = tid; i < DMOD; i += 256) { float v = xr[i]; s += v * v; }
  __shared__ float red[256];
  red[tid] = s; __syncthreads();
  for (int st = 128; st > 0; st >>= 1) {
    if (tid < st) red[tid] += red[tid + st];
    __syncthreads();
  }
  float inv = rsqrtf(red[0] * (1.0f / DMOD) + 1e-6f);
#pragma unroll
  for (int i = tid; i < DMOD; i += 256)
    y[(size_t)row * DMOD + i] = (bf16)(w[i] * xr[i] * inv);
}

// ---------------------------------------------------------------------------
// Tiled bf16 WMMA GEMM: C[M][N] = A[M][K] @ Bt[N][K]^T   (both bf16 row-major)
// Block tile 128x128, K-tile 64, double-buffered LDS (one barrier per tile);
// 8 waves, each 32x64 (2x4 wmma frags, 16 wmma per K-tile).
// Epilogues: 0=f32, 1=sigmoid(x+bias[n]) f32, 2=silu bf16, 3=x+res[m][n] f32
// ---------------------------------------------------------------------------
#define EPI_F32   0
#define EPI_SIG   1
#define EPI_SILU  2
#define EPI_ADD   3

#define BKT  64
#define LDST 72   // padded LDS row stride (bf16): 144B = 9*16B, conflict-free frags

template <int EPI>
__global__ __launch_bounds__(256) void gemm_bf16_kernel(
    const bf16* __restrict__ A, const bf16* __restrict__ Bt,
    void* __restrict__ Cout, const float* __restrict__ aux,
    int M, int N, int K) {
  __shared__ bf16 As[2][128 * LDST];
  __shared__ bf16 Bs[2][128 * LDST];
  int tid = threadIdx.x, lane = tid & 31, wave = tid >> 5;
  int bm = blockIdx.y * 128, bn = blockIdx.x * 128;
  int wm = (wave >> 1) * 32, wn = (wave & 1) * 64;

  f32x8 acc[2][4] = {};

  // per-thread staging: 128x64 bf16 tile = 1024 uint4, 4 per thread per matrix
  uint4 ra[4], rb[4];
  auto load_tile = [&](int k0) {
#pragma unroll
    for (int j = 0; j < 4; ++j) {
      int i = tid + j * 256, r = i >> 3, c = (i & 7) << 3;
      ra[j] = *(const uint4*)(A  + (size_t)(bm + r) * K + k0 + c);
      rb[j] = *(const uint4*)(Bt + (size_t)(bn + r) * K + k0 + c);
    }
  };
  auto store_tile = [&](int buf) {
#pragma unroll
    for (int j = 0; j < 4; ++j) {
      int i = tid + j * 256, r = i >> 3, c = (i & 7) << 3;
      *(uint4*)(&As[buf][r * LDST + c]) = ra[j];
      *(uint4*)(&Bs[buf][r * LDST + c]) = rb[j];
    }
  };

  load_tile(0);
  store_tile(0);
  __syncthreads();

  int nTiles = K / BKT;
  for (int ti = 0; ti < nTiles; ++ti) {
    int buf = ti & 1;
    bool hasNext = (ti + 1) < nTiles;
    if (hasNext) load_tile((ti + 1) * BKT);     // overlap global with compute

#pragma unroll
    for (int kk = 0; kk < 2; ++kk) {            // two 32-K slabs per tile
      bf16x16 af[2], bfm[4];
      af[0] = load_a_frag(&As[buf][(wm)      * LDST + kk * 32], LDST, lane);
      af[1] = load_a_frag(&As[buf][(wm + 16) * LDST + kk * 32], LDST, lane);
#pragma unroll
      for (int nt = 0; nt < 4; ++nt)
        bfm[nt] = load_b_frag(&Bs[buf][(wn + nt * 16) * LDST + kk * 32], LDST, lane);
#pragma unroll
      for (int mt = 0; mt < 2; ++mt)
#pragma unroll
        for (int nt = 0; nt < 4; ++nt)
          acc[mt][nt] = WMMA_BF16(af[mt], bfm[nt], acc[mt][nt]);
    }

    if (hasNext) store_tile(buf ^ 1);           // fill idle buffer
    __syncthreads();                            // one barrier per K-tile
  }

  int hi = lane >> 4, ln = lane & 15;
#pragma unroll
  for (int mt = 0; mt < 2; ++mt)
#pragma unroll
    for (int nt = 0; nt < 4; ++nt)
#pragma unroll
      for (int r = 0; r < 8; ++r) {
        int row = bm + wm + mt * 16 + r + hi * 8;
        int col = bn + wn + nt * 16 + ln;
        size_t idx = (size_t)row * N + col;
        float v = acc[mt][nt][r];
        if (EPI == EPI_F32) {
          ((float*)Cout)[idx] = v;
        } else if (EPI == EPI_SIG) {
          v += aux[col];
          ((float*)Cout)[idx] = 1.0f / (1.0f + __expf(-v));
        } else if (EPI == EPI_SILU) {
          ((bf16*)Cout)[idx] = (bf16)(v / (1.0f + __expf(-v)));
        } else { // EPI_ADD
          ((float*)Cout)[idx] = v + aux[idx];
        }
      }
}

// ---------------------------------------------------------------------------
// RoPE: q_f32[M][2048], k_f32[M][512] -> Q[b][h][t][d], K[b][kv][t][d] (bf16)
// position_ids = arange(B*T); interleaved-pair variant of the reference.
// ---------------------------------------------------------------------------
__global__ __launch_bounds__(256) void rope_kernel(
    const float* __restrict__ qf, const float* __restrict__ kf,
    const int* __restrict__ pos_ids,
    bf16* __restrict__ Q, bf16* __restrict__ Kr) {
  int bt = blockIdx.x, tid = threadIdx.x;
  int b = bt / TSEQ, t = bt - b * TSEQ;
  float pos = (float)pos_ids[bt];
  const float ln_base = 9.210340371976184f;          // log(10000)
  for (int p = tid; p < (NH + NKV) * (HD / 2); p += 256) {
    bool isQ = p < NH * (HD / 2);
    int pp   = isQ ? p : p - NH * (HD / 2);
    int head = pp >> 6, i = pp & 63;
    int d0 = 2 * i, d1 = 2 * i + 1;
    float a0 = pos * __expf(-ln_base * (float)(d0 & 63) * (1.0f / 64.0f));
    float a1 = pos * __expf(-ln_base * (float)(d1 & 63) * (1.0f / 64.0f));
    float x0, x1;
    size_t oidx;
    if (isQ) {
      const float* src = qf + (size_t)bt * (NH * HD) + head * HD;
      x0 = src[d0]; x1 = src[d1];
      oidx = (((size_t)b * NH + head) * TSEQ + t) * HD;
      Q[oidx + d0] = (bf16)(x0 * cosf(a0) - x1 * sinf(a0));
      Q[oidx + d1] = (bf16)(x1 * cosf(a1) + x0 * sinf(a1));
    } else {
      const float* src = kf + (size_t)bt * (NKV * HD) + head * HD;
      x0 = src[d0]; x1 = src[d1];
      oidx = (((size_t)b * NKV + head) * TSEQ + t) * HD;
      Kr[oidx + d0] = (bf16)(x0 * cosf(a0) - x1 * sinf(a0));
      Kr[oidx + d1] = (bf16)(x1 * cosf(a1) + x0 * sinf(a1));
    }
  }
}

// v_f32[M][512] -> Vt[b][kv][d][t] bf16  (K-contiguous for PV B-fragments)
__global__ __launch_bounds__(256) void vtrans_kernel(
    const float* __restrict__ vf, bf16* __restrict__ Vt) {
  int t  = blockIdx.x * 256 + threadIdx.x;
  int d  = blockIdx.y;
  int bk = blockIdx.z;                  // b*NKV + kv
  int b = bk >> 2, kv = bk & 3;
  float v = vf[((size_t)(b * TSEQ + t)) * (NKV * HD) + kv * HD + d];
  Vt[((size_t)bk * HD + d) * TSEQ + t] = (bf16)v;
}

// ---------------------------------------------------------------------------
// Flash attention (causal, GQA 16->4, hd=128), sigmoid-gate fused.
// Block = (128 queries, one head, one batch); 8 waves x 16 query rows.
// S = Q K^T and O += P V through WMMA; Q fragments register-resident across
// all key tiles; P relayout via per-wave LDS staging.
// ---------------------------------------------------------------------------
#define QLD 136   // Q/K LDS stride (272B = 17*16B) -> conflict-free frag loads
#define VLD 72    // V/P LDS stride (144B = 9*16B)

__global__ __launch_bounds__(256) void attn_kernel(
    const bf16* __restrict__ Q,    // [B][H][T][128]
    const bf16* __restrict__ Kr,   // [B][KV][T][128]
    const bf16* __restrict__ Vt,   // [B][KV][128][T]
    const float* __restrict__ gate,// [B*T][2048] (sigmoid already applied)
    bf16* __restrict__ Og) {       // [B*T][2048]
  __shared__ bf16 Qs[128 * QLD];
  __shared__ bf16 Ks[64 * QLD];
  __shared__ bf16 Vs[128 * VLD];
  __shared__ bf16 Ps[8 * 16 * VLD];

  int tid = threadIdx.x, lane = tid & 31, wave = tid >> 5;
  int hi = lane >> 4, ln = lane & 15;
  int qBase = blockIdx.x * 128, h = blockIdx.y, b = blockIdx.z;
  int kv = h >> 2;
  int qRow0 = qBase + wave * 16;

  const bf16* Qg = Q  + (((size_t)b * NH  + h ) * TSEQ + qBase) * HD;
  const bf16* Kg = Kr + (((size_t)b * NKV + kv) * TSEQ) * HD;
  const bf16* Vg = Vt + (((size_t)b * NKV + kv) * HD) * TSEQ;

  for (int i = tid; i < 2048; i += 256) {     // 128 x 128 bf16 Q tile
    int r = i >> 4, c = (i & 15) << 3;
    *(uint4*)(Qs + r * QLD + c) = *(const uint4*)(Qg + (size_t)r * HD + c);
  }
  __syncthreads();

  // Q A-fragments are invariant across key tiles: hoist into registers.
  bf16x16 qa[4];
#pragma unroll
  for (int ks = 0; ks < 4; ++ks)
    qa[ks] = load_a_frag(Qs + (wave * 16) * QLD + ks * 32, QLD, lane);

  f32x8 accO[8] = {};
  float mst[8], lst[8];
#pragma unroll
  for (int r = 0; r < 8; ++r) { mst[r] = -1e30f; lst[r] = 0.f; }

  const float sc = 0.08838834764831845f;      // 1/sqrt(128)
  int nkt = (qBase + 128) >> 6;               // causal: skip tiles past diagonal

  for (int kt = 0; kt < nkt; ++kt) {
    int kb = kt << 6;
    __syncthreads();
    for (int i = tid; i < 1024; i += 256) {   // K tile: 64 keys x 128 dims
      int r = i >> 4, c = (i & 15) << 3;
      *(uint4*)(Ks + r * QLD + c) = *(const uint4*)(Kg + (size_t)(kb + r) * HD + c);
    }
    for (int i = tid; i < 1024; i += 256) {   // V^T tile: 128 dims x 64 keys
      int r = i >> 3, c = (i & 7) << 3;
      *(uint4*)(Vs + r * VLD + c) = *(const uint4*)(Vg + (size_t)r * TSEQ + kb + c);
    }
    __syncthreads();

    // ---- S = Q K^T (16 x 64 per wave); Q frags already in registers ----
    f32x8 s[4] = {};
#pragma unroll
    for (int ks = 0; ks < 4; ++ks) {
      bf16x16 kb0[4];
#pragma unroll
      for (int nt = 0; nt < 4; ++nt)          // batch the 8 ds_loads per slab
        kb0[nt] = load_b_frag(Ks + (nt * 16) * QLD + ks * 32, QLD, lane);
#pragma unroll
      for (int nt = 0; nt < 4; ++nt)
        s[nt] = WMMA_BF16(qa[ks], kb0[nt], s[nt]);
    }

    // ---- scale, causal mask, online softmax ----
    bool needMask = (kb + 63) > qRow0;
    float mloc[8];
#pragma unroll
    for (int r = 0; r < 8; ++r) mloc[r] = -1e30f;
#pragma unroll
    for (int nt = 0; nt < 4; ++nt)
#pragma unroll
      for (int r = 0; r < 8; ++r) {
        float v = s[nt][r] * sc;
        if (needMask) {
          int row = qRow0 + r + hi * 8;
          int col = kb + nt * 16 + ln;
          if (col > row) v = -1e9f;
        }
        s[nt][r] = v;
        mloc[r] = fmaxf(mloc[r], v);
      }
#pragma unroll
    for (int off = 1; off < 16; off <<= 1)
#pragma unroll
      for (int r = 0; r < 8; ++r)
        mloc[r] = fmaxf(mloc[r], __shfl_xor(mloc[r], off, 32));

    float scl[8];
#pragma unroll
    for (int r = 0; r < 8; ++r) {
      float mn = fmaxf(mst[r], mloc[r]);
      scl[r] = __expf(mst[r] - mn);
      mst[r] = mn;
      lst[r] *= scl[r];
    }

    bf16* Pw = Ps + wave * 16 * VLD;
    float rs[8];
#pragma unroll
    for (int r = 0; r < 8; ++r) rs[r] = 0.f;
#pragma unroll
    for (int nt = 0; nt < 4; ++nt)
#pragma unroll
      for (int r = 0; r < 8; ++r) {
        float p = __expf(s[nt][r] - mst[r]);
        rs[r] += p;
        Pw[(r + hi * 8) * VLD + nt * 16 + ln] = (bf16)p;
      }
#pragma unroll
    for (int off = 1; off < 16; off <<= 1)
#pragma unroll
      for (int r = 0; r < 8; ++r)
        rs[r] += __shfl_xor(rs[r], off, 32);
#pragma unroll
    for (int r = 0; r < 8; ++r) lst[r] += rs[r];
#pragma unroll
    for (int dt = 0; dt < 8; ++dt)
#pragma unroll
      for (int r = 0; r < 8; ++r) accO[dt][r] *= scl[r];

    // LDS pipe is in-order within a wave; fence compiler + wait DS writes.
    asm volatile("s_wait_dscnt 0" ::: "memory");

    // ---- O += P V ----
#pragma unroll
    for (int ks2 = 0; ks2 < 2; ++ks2) {
      bf16x16 a = load_a_frag(Pw + ks2 * 32, VLD, lane);
#pragma unroll
      for (int dt = 0; dt < 8; ++dt) {
        bf16x16 bb = load_b_frag(Vs + (dt * 16) * VLD + ks2 * 32, VLD, lane);
        accO[dt] = WMMA_BF16(a, bb, accO[dt]);
      }
    }
  }

  // ---- normalize, gate, store bf16 [b*T][H*128] ----
#pragma unroll
  for (int dt = 0; dt < 8; ++dt)
#pragma unroll
    for (int r = 0; r < 8; ++r) {
      int t = qRow0 + r + hi * 8;
      int col = h * HD + dt * 16 + ln;
      size_t idx = ((size_t)(b * TSEQ + t)) * (NH * HD) + col;
      float o = (accO[dt][r] / lst[r]) * gate[idx];
      Og[idx] = (bf16)o;
    }
}

// ---------------------------------------------------------------------------
// Host orchestration
// ---------------------------------------------------------------------------
extern "C" void kernel_launch(void* const* d_in, const int* in_sizes, int n_in,
                              void* d_out, int out_size, void* d_ws, size_t ws_size,
                              hipStream_t stream) {
  const float* hidden      = (const float*)d_in[0];
  // d_in[1]: attention_mask (pure causal; applied analytically)
  const int*   pos_ids     = (const int*)d_in[2];
  const float* attn_norm_w = (const float*)d_in[3];
  const float* mlp_norm_w  = (const float*)d_in[4];
  const float* wq   = (const float*)d_in[5];
  const float* wk   = (const float*)d_in[6];
  const float* wv   = (const float*)d_in[7];
  const float* wo   = (const float*)d_in[8];
  const float* gw   = (const float*)d_in[9];
  const float* gb   = (const float*)d_in[10];
  const float* fc1  = (const float*)d_in[11];
  const float* fc2  = (const float*)d_in[12];
  float* out = (float*)d_out;

  char* ws = (char*)d_ws;
  size_t off = 0;
  auto alloc = [&](size_t bytes) { size_t o = off; off += (bytes + 255) & ~(size_t)255; return o; };

  bf16*  wqT   = (bf16*)(ws + alloc((size_t)DMOD * DMOD * 2));
  bf16*  wkT   = (bf16*)(ws + alloc((size_t)(NKV * HD) * DMOD * 2));
  bf16*  wvT   = (bf16*)(ws + alloc((size_t)(NKV * HD) * DMOD * 2));
  bf16*  woT   = (bf16*)(ws + alloc((size_t)DMOD * DMOD * 2));
  bf16*  gwT   = (bf16*)(ws + alloc((size_t)DMOD * DMOD * 2));
  bf16*  fc1T  = (bf16*)(ws + alloc((size_t)INTER * DMOD * 2));
  bf16*  fc2T  = (bf16*)(ws + alloc((size_t)DMOD * INTER * 2));
  bf16*  hbf   = (bf16*)(ws + alloc((size_t)MTOK * DMOD * 2));
  float* qf    = (float*)(ws + alloc((size_t)MTOK * DMOD * 4));   // reused as res2
  float* kf    = (float*)(ws + alloc((size_t)MTOK * NKV * HD * 4));
  float* vf    = (float*)(ws + alloc((size_t)MTOK * NKV * HD * 4));
  float* gatef = (float*)(ws + alloc((size_t)MTOK * DMOD * 4));   // reused as h2
  bf16*  Qr    = (bf16*)(ws + alloc((size_t)MTOK * NH * HD * 2));
  bf16*  Kr    = (bf16*)(ws + alloc((size_t)MTOK * NKV * HD * 2));
  bf16*  Vt    = (bf16*)(ws + alloc((size_t)MTOK * NKV * HD * 2));
  bf16*  attnG = (bf16*)(ws + alloc((size_t)MTOK * DMOD * 2));
  bf16*  m1    = (bf16*)(ws + alloc((size_t)MTOK * INTER * 2));
  float* res2  = qf;                    // q_f32 dead after RoPE
  bf16*  h2    = (bf16*)gatef;          // gate dead after attention

  dim3 tb(256);

  // 1) weights -> bf16 transposed
  transpose_cvt_kernel<<<dim3(DMOD / 32, DMOD / 32), tb, 0, stream>>>(wq, wqT, DMOD, DMOD);
  transpose_cvt_kernel<<<dim3((NKV * HD) / 32, DMOD / 32), tb, 0, stream>>>(wk, wkT, DMOD, NKV * HD);
  transpose_cvt_kernel<<<dim3((NKV * HD) / 32, DMOD / 32), tb, 0, stream>>>(wv, wvT, DMOD, NKV * HD);
  transpose_cvt_kernel<<<dim3(DMOD / 32, DMOD / 32), tb, 0, stream>>>(wo, woT, DMOD, DMOD);
  transpose_cvt_kernel<<<dim3(DMOD / 32, DMOD / 32), tb, 0, stream>>>(gw, gwT, DMOD, DMOD);
  transpose_cvt_kernel<<<dim3(INTER / 32, DMOD / 32), tb, 0, stream>>>(fc1, fc1T, DMOD, INTER);
  transpose_cvt_kernel<<<dim3(DMOD / 32, INTER / 32), tb, 0, stream>>>(fc2, fc2T, INTER, DMOD);

  // 2) attn RMSNorm
  rmsnorm_kernel<<<MTOK, tb, 0, stream>>>(hidden, attn_norm_w, hbf);

  // 3) projections
  gemm_bf16_kernel<EPI_F32><<<dim3(DMOD / 128, MTOK / 128), tb, 0, stream>>>(
      hbf, wqT, qf, nullptr, MTOK, DMOD, DMOD);
  gemm_bf16_kernel<EPI_F32><<<dim3((NKV * HD) / 128, MTOK / 128), tb, 0, stream>>>(
      hbf, wkT, kf, nullptr, MTOK, NKV * HD, DMOD);
  gemm_bf16_kernel<EPI_F32><<<dim3((NKV * HD) / 128, MTOK / 128), tb, 0, stream>>>(
      hbf, wvT, vf, nullptr, MTOK, NKV * HD, DMOD);
  gemm_bf16_kernel<EPI_SIG><<<dim3(DMOD / 128, MTOK / 128), tb, 0, stream>>>(
      hbf, gwT, gatef, gb, MTOK, DMOD, DMOD);

  // 4) RoPE + V transpose
  rope_kernel<<<MTOK, tb, 0, stream>>>(qf, kf, pos_ids, Qr, Kr);
  vtrans_kernel<<<dim3(TSEQ / 256, HD, BSZ * NKV), tb, 0, stream>>>(vf, Vt);

  // 5) attention (causal, gated)
  attn_kernel<<<dim3(TSEQ / 128, NH, BSZ), tb, 0, stream>>>(Qr, Kr, Vt, gatef, attnG);

  // 6) output projection + residual
  gemm_bf16_kernel<EPI_ADD><<<dim3(DMOD / 128, MTOK / 128), tb, 0, stream>>>(
      attnG, woT, res2, hidden, MTOK, DMOD, DMOD);

  // 7) MLP
  rmsnorm_kernel<<<MTOK, tb, 0, stream>>>(res2, mlp_norm_w, h2);
  gemm_bf16_kernel<EPI_SILU><<<dim3(INTER / 128, MTOK / 128), tb, 0, stream>>>(
      h2, fc1T, m1, nullptr, MTOK, INTER, DMOD);
  gemm_bf16_kernel<EPI_ADD><<<dim3(DMOD / 128, MTOK / 128), tb, 0, stream>>>(
      m1, fc2T, out, res2, MTOK, DMOD, INTER);
}
